// GCNMambaFusion_17325898072385
// MI455X (gfx1250) — compile-verified, matrix operand order checked by reading, not run on hardware
//
#include <hip/hip_runtime.h>

// ---------------------------------------------------------------------------
// CDNA5 (gfx1250) implementation of GCN+Mamba fusion.
// WMMA bf16 16x16x32 for all large GEMMs (LDS-staged weights, A-frag reuse),
// fused similarity-GEMM + streaming top-9, scalar kernels for scans/norms.
// ---------------------------------------------------------------------------

typedef __attribute__((ext_vector_type(16))) __bf16 v16bf;
typedef __attribute__((ext_vector_type(8)))  __bf16 v8bf;
typedef __attribute__((ext_vector_type(8)))  float  v8f;
typedef __attribute__((ext_vector_type(4)))  int    i32x4;

#define CDIV(a,b) (((a)+(b)-1)/(b))

// Optional gfx1250 async global->LDS staging (guarded; falls back to
// global_load + ds_store when the builtins are not declared).
// Probe result: the builtin's pointer params are 'int4*' (16B payload).
#if defined(__has_builtin)
# if __has_builtin(__builtin_amdgcn_global_load_async_to_lds_b128) && \
     __has_builtin(__builtin_amdgcn_s_wait_asynccnt)
#  define GEMM_ASYNC 1
# endif
#endif
#ifndef GEMM_ASYNC
# define GEMM_ASYNC 0
#endif

static constexpr int Bc = 32, Cc = 64, Hh = 32, Ww = 32, Nn = 1024; // N = H*W
static constexpr int KTOP = 9;

__device__ __forceinline__ float sigmoidf_(float x){ return 1.f/(1.f+__expf(-x)); }
__device__ __forceinline__ float siluf_(float x){ return x*sigmoidf_(x); }

// ---- WMMA helpers ----------------------------------------------------------
// 16-bit A/B fragment, 16x32 (MxK) layout per ISA 7.12.2:
//   lanes 0-15:  row = lane,    K = {0..7, 16..23}
//   lanes 16-31: row = lane-16, K = {8..15, 24..31}
// p must point at row_base + k0 + hiHalf*8 (16B aligned).
__device__ __forceinline__ v16bf load_rowK(const __bf16* p){
  v8bf lo = *(const v8bf*)p;
  v8bf hi = *(const v8bf*)(p + 16);
  v16bf a;
#pragma unroll
  for (int e = 0; e < 8; ++e){ a[e] = lo[e]; a[e+8] = hi[e]; }
  return a;
}

__device__ __forceinline__ v8f wmma_bf16(v16bf a, v16bf b, v8f c){
  return __builtin_amdgcn_wmma_f32_16x16x32_bf16(false, a, false, b,
                                                 (short)0, c, false, false);
}

// ---- K1: row-normalize x (B,C,HW) -> xn (B,HW,C) f32 + bf16 ---------------
__global__ __launch_bounds__(64)
void k_normalize(const float* __restrict__ X, float* __restrict__ XN,
                 __bf16* __restrict__ XN16){
  int blk = blockIdx.x;
  int b = blk >> 10, n = blk & 1023;
  int c = threadIdx.x;
  float v = X[((size_t)b*Cc + c)*Nn + n];
  __shared__ float red[64];
  red[c] = v*v; __syncthreads();
  for (int s = 32; s > 0; s >>= 1){ if (c < s) red[c] += red[c+s]; __syncthreads(); }
  float inv = 1.f / fmaxf(sqrtf(red[0]), 1e-12f);
  float xv = v * inv;
  size_t o = ((size_t)b*Nn + n)*Cc + c;
  XN[o] = xv; XN16[o] = (__bf16)xv;
}

// ---- K2: fused similarity GEMM (WMMA) + streaming top-9 -------------------
// One wave per 16-row block of sim; 64 column tiles x 2 WMMAs streamed, sim
// never hits HBM. Both half-waves scan (8 cols each), early-out threshold,
// merge of the two half-wave top-9 lists through LDS at the end.
__global__ __launch_bounds__(32)
void k_sim_topk(const __bf16* __restrict__ XN16, float* __restrict__ VAL,
                int* __restrict__ IDX){
  int b  = blockIdx.x >> 6;
  int m0 = (blockIdx.x & 63) << 4;
  int lane = threadIdx.x & 31;
  int r = lane & 15, hih = lane >> 4;

  const __bf16* base = XN16 + (size_t)b*Nn*Cc;
  v16bf a0 = load_rowK(base + (size_t)(m0 + r)*Cc +  0 + hih*8);
  v16bf a1 = load_rowK(base + (size_t)(m0 + r)*Cc + 32 + hih*8);

  __shared__ float smem[16][17];
  __shared__ float mv[16][KTOP];
  __shared__ int   mi[16][KTOP];
  float tv[KTOP]; int ti[KTOP];
#pragma unroll
  for (int k = 0; k < KTOP; ++k){ tv[k] = -1e30f; ti[k] = 0; }

  const int row = r, jb = hih * 8;      // each half-wave scans 8 columns
  for (int n0 = 0; n0 < Nn; n0 += 16){
    if (n0 + 16 < Nn)
      __builtin_prefetch(base + (size_t)(n0 + 16 + r)*Cc, 0, 1);
    v16bf b0 = load_rowK(base + (size_t)(n0 + r)*Cc +  0 + hih*8);
    v16bf b1 = load_rowK(base + (size_t)(n0 + r)*Cc + 32 + hih*8);
    v8f acc = {};
    acc = wmma_bf16(a0, b0, acc);
    acc = wmma_bf16(a1, b1, acc);
    // C layout: lane -> col N = lane&15; VGPR q -> row M = q + 8*hiHalf
#pragma unroll
    for (int q = 0; q < 8; ++q) smem[jb + q][r] = acc[q];
    __syncthreads();
#pragma unroll
    for (int j = 0; j < 8; ++j){
      float v = smem[row][jb + j]; int id = n0 + jb + j;
      if (v > tv[KTOP-1]){                        // early-out: rare to pass
#pragma unroll
        for (int p = 0; p < KTOP; ++p){
          if (v > tv[p]){ float fv = tv[p]; int fi = ti[p];
                          tv[p] = v; ti[p] = id; v = fv; id = fi; }
        }
      }
    }
    __syncthreads();
  }
  // merge the two half-wave lists
  if (lane >= 16){
#pragma unroll
    for (int k = 0; k < KTOP; ++k){ mv[row][k] = tv[k]; mi[row][k] = ti[k]; }
  }
  __syncthreads();
  if (lane < 16){
#pragma unroll
    for (int k = 0; k < KTOP; ++k){
      float v = mv[row][k]; int id = mi[row][k];
      if (v > tv[KTOP-1]){
#pragma unroll
        for (int p = 0; p < KTOP; ++p){
          if (v > tv[p]){ float fv = tv[p]; int fi = ti[p];
                          tv[p] = v; ti[p] = id; v = fv; id = fi; }
        }
      }
    }
    float s = 1e-6f;
#pragma unroll
    for (int k = 0; k < KTOP; ++k) s += tv[k];
    float inv = 1.f / s;
    size_t o = ((size_t)b*Nn + m0 + row)*KTOP;
#pragma unroll
    for (int k = 0; k < KTOP; ++k){ VAL[o+k] = tv[k]*inv; IDX[o+k] = ti[k]; }
  }
}

// ---- generic WMMA GEMM: Y[M,Npad] = X[M,K](bf16) @ op(W)(bf16) ------------
// 4 waves / block, M-block = 64 rows. Whole weight matrix staged in LDS as
// [n][k] (zero-padded past Nvalid); A-fragments loaded once per wave into
// registers and reused across all N tiles. transW=1: W is (N,K) row-major;
// transW=0: W is (K,N) row-major (transposed while staging).
template<int K>
__global__ __launch_bounds__(128)
void k_gemm_wmma(const __bf16* __restrict__ X, const __bf16* __restrict__ Wm,
                 float* __restrict__ Y, int Ntiles, int Nvalid,
                 int ldw, int ldy, int transW){
  extern __shared__ __bf16 ldsW[];
  const int tid = threadIdx.x;
  const int totalE = Ntiles*16*K;
  if (transW){
    for (int i = tid*8; i < totalE; i += 128*8){
      int n = i / K, k = i % K;
      if (n < Nvalid){
#if GEMM_ASYNC
        // async copy: 16B per lane, LDS dest first (ISA: VDST = LDS addr)
        __builtin_amdgcn_global_load_async_to_lds_b128(
            (i32x4*)(ldsW + i),
            (i32x4*)(size_t)(Wm + (size_t)n*ldw + k),
            0, 0);
#else
        *(v8bf*)(ldsW + i) = *(const v8bf*)(Wm + (size_t)n*ldw + k);
#endif
      } else {
        v8bf z;
#pragma unroll
        for (int e = 0; e < 8; ++e) z[e] = (__bf16)0.f;
        *(v8bf*)(ldsW + i) = z;
      }
    }
#if GEMM_ASYNC
    __builtin_amdgcn_s_wait_asynccnt(0);
#endif
  } else {
    for (int i = tid; i < totalE; i += 128){
      int n = i / K, k = i % K;
      ldsW[i] = (n < Nvalid) ? Wm[(size_t)k*ldw + n] : (__bf16)0.f;
    }
  }
  __syncthreads();

  const int lane = tid & 31, wave = tid >> 5;
  const int r = lane & 15, hih = lane >> 4;
  const int m0 = blockIdx.x*64 + wave*16;
  constexpr int NK = K/32;
  v16bf afr[NK];
#pragma unroll
  for (int kk = 0; kk < NK; ++kk)
    afr[kk] = load_rowK(X + (size_t)(m0 + r)*K + kk*32 + hih*8);
  const int mb = hih*8;
  for (int n0 = 0; n0 < Ntiles*16; n0 += 16){
    v8f acc = {};
#pragma unroll
    for (int kk = 0; kk < NK; ++kk){
      v16bf bf = load_rowK(ldsW + (size_t)(n0 + r)*K + kk*32 + hih*8);
      acc = wmma_bf16(afr[kk], bf, acc);
    }
#pragma unroll
    for (int q = 0; q < 8; ++q)
      if (n0 + r < Nvalid) Y[(size_t)(m0 + mb + q)*ldy + n0 + r] = acc[q];
  }
}

// ---- GCN neighbor aggregation ---------------------------------------------
__global__ __launch_bounds__(256)
void k_gather(const float* __restrict__ XT, const float* __restrict__ VAL,
              const int* __restrict__ IDX, const float* __restrict__ bg,
              float* __restrict__ O){
  int t = blockIdx.x*256 + threadIdx.x;
  int c = t & 63, n = (t >> 6) & 1023, b = t >> 16;
  size_t vb = ((size_t)b*Nn + n)*KTOP;
  float acc = bg[c];
#pragma unroll
  for (int k = 0; k < KTOP; ++k)
    acc += VAL[vb+k] * XT[((size_t)b*Nn + IDX[vb+k])*Cc + c];
  O[(size_t)t] = acc;
}

// ---- group-norm (4 groups) + SiLU; output (B,C,HW) f32 --------------------
__device__ __forceinline__ float gn_load(const float* in, int mode, int b,
                                         int c, int n){
  if (mode == 0) return in[((size_t)b*Nn + n)*Cc + c];            // (B,HW,C)
  if (mode == 2){ int h = n >> 5, w = n & 31;                     // (B,W,H,C)
                  return in[(((size_t)b*Ww + w)*Hh + h)*Cc + c]; }
  return in[((size_t)b*Cc + c)*Nn + n];                           // (B,C,HW)
}

__global__ __launch_bounds__(256)
void k_groupnorm_silu(const float* __restrict__ in, const float* __restrict__ g,
                      const float* __restrict__ be, float* __restrict__ out,
                      int mode){
  int b = blockIdx.x >> 2, grp = blockIdx.x & 3;
  const int CNT = 16 * Nn;
  float s = 0.f, s2 = 0.f;
  for (int i = threadIdx.x; i < CNT; i += 256){
    int c = grp*16 + (i >> 10), n = i & 1023;
    float v = gn_load(in, mode, b, c, n);
    s += v; s2 += v*v;
  }
  __shared__ float rs[256], rq[256];
  rs[threadIdx.x] = s; rq[threadIdx.x] = s2; __syncthreads();
  for (int st = 128; st > 0; st >>= 1){
    if (threadIdx.x < st){ rs[threadIdx.x] += rs[threadIdx.x+st];
                           rq[threadIdx.x] += rq[threadIdx.x+st]; }
    __syncthreads();
  }
  float mean = rs[0] / CNT;
  float inv  = rsqrtf(rq[0]/CNT - mean*mean + 1e-5f);
  for (int i = threadIdx.x; i < CNT; i += 256){
    int c = grp*16 + (i >> 10), n = i & 1023;
    float v = gn_load(in, mode, b, c, n);
    float y = (v - mean)*inv*g[c] + be[c];
    out[((size_t)b*Cc + c)*Nn + n] = siluf_(y);
  }
}

// ---- layout/convert helpers -----------------------------------------------
__global__ __launch_bounds__(256)
void k_f2bf(const float* __restrict__ s, __bf16* __restrict__ d, int n){
  int t = blockIdx.x*256 + threadIdx.x;
  if (t < n) d[t] = (__bf16)s[t];
}
__global__ __launch_bounds__(256)
void k_bcn2bnc_bf16(const float* __restrict__ s, __bf16* __restrict__ d){
  int t = blockIdx.x*256 + threadIdx.x;              // dst idx (b,n,c)
  int c = t & 63, n = (t >> 6) & 1023, b = t >> 16;
  d[t] = (__bf16)s[((size_t)b*Cc + c)*Nn + n];
}
__global__ __launch_bounds__(256)
void k_row2col_bf16(const float* __restrict__ s, __bf16* __restrict__ d){
  int t = blockIdx.x*256 + threadIdx.x;              // dst idx (b,w,h,c)
  int c = t & 63, h = (t >> 6) & 31, w = (t >> 11) & 31, b = t >> 16;
  d[t] = (__bf16)s[(((size_t)b*Hh + h)*Ww + w)*Cc + c];
}

// ---- mamba stage kernels ---------------------------------------------------
__global__ __launch_bounds__(256)
void k_conv_silu(const float* __restrict__ XZ, const float* __restrict__ cw,
                 const float* __restrict__ cb, float* __restrict__ XC,
                 __bf16* __restrict__ XC16, int S, int L, int di, int ldxz){
  int t = blockIdx.x*256 + threadIdx.x;
  if (t >= S*L*di) return;
  int c = t % di; int m = t / di; int l = m % L; int sq = m / L;
  float acc = cb[c];
#pragma unroll
  for (int j = 0; j < 4; ++j){
    int ll = l - 3 + j;
    if (ll >= 0) acc += cw[c*4 + j] * XZ[(size_t)(sq*L + ll)*ldxz + c];
  }
  float y = siluf_(acc);
  XC[(size_t)m*di + c] = y;
  XC16[(size_t)m*di + c] = (__bf16)y;
}

__global__ __launch_bounds__(256)
void k_dt(const float* __restrict__ DBC, const float* __restrict__ Wdt,
          const float* __restrict__ bdt, float* __restrict__ DT,
          long Mrows, int di, int r, int lddbc){
  long t = (long)blockIdx.x*256 + threadIdx.x;
  if (t >= Mrows*di) return;
  int c = (int)(t % di); long m = t / di;
  float x = bdt[c];
  for (int j = 0; j < r; ++j) x += DBC[(size_t)m*lddbc + j] * Wdt[c*r + j];
  DT[(size_t)m*di + c] = (x > 20.f) ? x : log1pf(__expf(x));
}

__global__ __launch_bounds__(256)
void k_scan(const float* __restrict__ DT, const float* __restrict__ XC,
            const float* __restrict__ DBC, const float* __restrict__ XZ,
            const float* __restrict__ A_log, const float* __restrict__ Dp,
            float* __restrict__ Yo, __bf16* __restrict__ Y16,
            int S, int L, int di, int lddbc, int roff, int ldxz){
  int t = blockIdx.x*256 + threadIdx.x;
  if (t >= S*di) return;
  int c = t % di, sq = t / di;
  float negA[16], h[16];
#pragma unroll
  for (int n = 0; n < 16; ++n){ negA[n] = -__expf(A_log[c*16 + n]); h[n] = 0.f; }
  const float* Dpc = Dp + c;
  for (int l = 0; l < L; ++l){
    size_t m = (size_t)sq*L + l;
    float dtv = DT[m*di + c];
    float xv  = XC[m*di + c];
    const float* bm = DBC + m*lddbc + roff;
    const float* cm = bm + 16;
    float du = dtv * xv, acc = 0.f;
#pragma unroll
    for (int n = 0; n < 16; ++n){
      h[n] = h[n]*__expf(dtv*negA[n]) + du*bm[n];
      acc += h[n]*cm[n];
    }
    float z = XZ[m*ldxz + di + c];
    float yv = (acc + xv*Dpc[0]) * siluf_(z);
    Yo[m*di + c] = yv; Y16[m*di + c] = (__bf16)yv;
  }
}

// ---- spectral (d_model=8) scalar kernels ----------------------------------
__global__ __launch_bounds__(256)
void k_sp_xz(const float* __restrict__ X, const float* __restrict__ pe,
             const float* __restrict__ Win, float* __restrict__ XZ){
  int t = blockIdx.x*256 + threadIdx.x;        // (m2, j) with j<32
  int j = t & 31; long m = t >> 5;
  if (m >= (long)Bc*Nn*8) return;
  int l = (int)(m & 7); long sq = m >> 3;
  int b = (int)(sq >> 10), n = (int)(sq & 1023);
  float acc = 0.f;
#pragma unroll
  for (int d = 0; d < 8; ++d){
    float xv = X[((size_t)b*Cc + l*8 + d)*Nn + n] + pe[d];
    acc += xv * Win[j*8 + d];
  }
  XZ[(size_t)m*32 + j] = acc;
}

__global__ __launch_bounds__(256)
void k_sgemm_xwT(const float* __restrict__ X, const float* __restrict__ Wm,
                 float* __restrict__ Y, long Mrows, int Nout, int Kd, int ldy){
  long t = (long)blockIdx.x*256 + threadIdx.x;
  if (t >= Mrows*Nout) return;
  int j = (int)(t % Nout); long m = t / Nout;
  float acc = 0.f;
  for (int k = 0; k < Kd; ++k) acc += X[(size_t)m*Kd + k]*Wm[j*Kd + k];
  Y[(size_t)m*ldy + j] = acc;
}

// ---- final fusion ----------------------------------------------------------
__global__ __launch_bounds__(256)
void k_fusion(const float* __restrict__ gcn, const float* __restrict__ spa,
              const float* __restrict__ spe, const float* __restrict__ x,
              const float* __restrict__ bw, const float* __restrict__ fw,
              float* __restrict__ out){
  int t = blockIdx.x*256 + threadIdx.x;
  float e0 = __expf(bw[0]), e1 = __expf(bw[1]);
  float bw0 = e0/(e0+e1), bw1 = e1/(e0+e1);
  float f0 = __expf(fw[0]), f1 = __expf(fw[1]);
  float fw0 = f0/(f0+f1), fw1 = f1/(f0+f1);
  out[t] = gcn[t]*fw0 + (spa[t]*bw0 + spe[t]*bw1)*fw1 + x[t];
}

// ---------------------------------------------------------------------------
extern "C" void kernel_launch(void* const* d_in, const int* in_sizes, int n_in,
                              void* d_out, int out_size, void* d_ws, size_t ws_size,
                              hipStream_t stream){
  (void)in_sizes; (void)n_in; (void)out_size; (void)ws_size;
  const float* x = (const float*)d_in[0];
  // row mamba
  const float* rWin = (const float*)d_in[1],  *rCw = (const float*)d_in[2];
  const float* rCb  = (const float*)d_in[3],  *rWx = (const float*)d_in[4];
  const float* rWdt = (const float*)d_in[5],  *rBdt= (const float*)d_in[6];
  const float* rAl  = (const float*)d_in[7],  *rD  = (const float*)d_in[8];
  const float* rWo  = (const float*)d_in[9];
  // col mamba
  const float* cWin = (const float*)d_in[10], *cCw = (const float*)d_in[11];
  const float* cCb  = (const float*)d_in[12], *cWx = (const float*)d_in[13];
  const float* cWdt = (const float*)d_in[14], *cBdt= (const float*)d_in[15];
  const float* cAl  = (const float*)d_in[16], *cD  = (const float*)d_in[17];
  const float* cWo  = (const float*)d_in[18];
  const float* spaG = (const float*)d_in[19], *spaB= (const float*)d_in[20];
  // spectral mamba
  const float* sWin = (const float*)d_in[21], *sCw = (const float*)d_in[22];
  const float* sCb  = (const float*)d_in[23], *sWx = (const float*)d_in[24];
  const float* sWdt = (const float*)d_in[25], *sBdt= (const float*)d_in[26];
  const float* sAl  = (const float*)d_in[27], *sD  = (const float*)d_in[28];
  const float* sWo  = (const float*)d_in[29], *sPe = (const float*)d_in[30];
  const float* speG = (const float*)d_in[31], *speB= (const float*)d_in[32];
  const float* bothW= (const float*)d_in[33];
  const float* gW1  = (const float*)d_in[34], *gB1 = (const float*)d_in[35];
  const float* gW2  = (const float*)d_in[36], *gB2 = (const float*)d_in[37];
  const float* gn1G = (const float*)d_in[38], *gn1B= (const float*)d_in[39];
  const float* gn2G = (const float*)d_in[40], *gn2B= (const float*)d_in[41];
  const float* fusW = (const float*)d_in[42];
  float* out = (float*)d_out;

  char* wsp = (char*)d_ws;
  auto carve = [&](size_t bytes)->char*{
    char* p = wsp; wsp += (bytes + 255) & ~(size_t)255; return p;
  };
  const size_t E = (size_t)Bc*Cc*Nn;                     // 2,097,152
  float*  XN    = (float*)  carve(E*4);
  __bf16* XN16  = (__bf16*) carve(E*2);
  __bf16* T16   = (__bf16*) carve(E*2);
  float*  VAL   = (float*)  carve((size_t)Bc*Nn*KTOP*4);
  int*    IDX   = (int*)    carve((size_t)Bc*Nn*KTOP*4);
  float*  T1    = (float*)  carve(E*4);
  float*  T2    = (float*)  carve(E*4);
  float*  GCN   = (float*)  carve(E*4);
  __bf16* MX16  = (__bf16*) carve(E*2);
  float*  XZ    = (float*)  carve((size_t)8388608*4);
  float*  XC    = (float*)  carve((size_t)4194304*4);
  __bf16* XC16  = (__bf16*) carve((size_t)4194304*2);
  float*  DBC   = (float*)  carve((size_t)8650752*4);
  float*  DT    = (float*)  carve((size_t)4194304*4);
  float*  Yb    = (float*)  carve((size_t)4194304*4);
  __bf16* Y16   = (__bf16*) carve((size_t)4194304*2);
  float*  MOUT  = (float*)  carve(E*4);
  __bf16* W1_16 = (__bf16*) carve(4096*2);
  __bf16* W2_16 = (__bf16*) carve(4096*2);
  __bf16* rWin16= (__bf16*) carve(16384*2);
  __bf16* rWx16 = (__bf16*) carve(4608*2);
  __bf16* rWo16 = (__bf16*) carve(8192*2);
  __bf16* cWin16= (__bf16*) carve(16384*2);
  __bf16* cWx16 = (__bf16*) carve(4608*2);
  __bf16* cWo16 = (__bf16*) carve(8192*2);

  const int M = Bc*Nn;                    // 32768 rows for row/col mamba + GCN
  // weight conversions
  k_f2bf<<<CDIV(4096,256),256,0,stream>>>(gW1, W1_16, 4096);
  k_f2bf<<<CDIV(4096,256),256,0,stream>>>(gW2, W2_16, 4096);
  k_f2bf<<<CDIV(16384,256),256,0,stream>>>(rWin, rWin16, 16384);
  k_f2bf<<<CDIV(4608,256),256,0,stream>>>(rWx, rWx16, 4608);
  k_f2bf<<<CDIV(8192,256),256,0,stream>>>(rWo, rWo16, 8192);
  k_f2bf<<<CDIV(16384,256),256,0,stream>>>(cWin, cWin16, 16384);
  k_f2bf<<<CDIV(4608,256),256,0,stream>>>(cWx, cWx16, 4608);
  k_f2bf<<<CDIV(8192,256),256,0,stream>>>(cWo, cWo16, 8192);

  // ---- GCN branch ----
  k_normalize<<<Bc*Nn, 64, 0, stream>>>(x, XN, XN16);
  k_sim_topk<<<Bc*(Nn/16), 32, 0, stream>>>(XN16, VAL, IDX);
  k_gemm_wmma<64><<<M/64, 128, 8192, stream>>>(XN16, W1_16, T1, 4, 64, 64, 64, 0);
  k_gather<<<CDIV((int)E,256),256,0,stream>>>(T1, VAL, IDX, gB1, T2);
  k_groupnorm_silu<<<Bc*4,256,0,stream>>>(T2, gn1G, gn1B, XN, 0);
  k_bcn2bnc_bf16<<<CDIV((int)E,256),256,0,stream>>>(XN, T16);
  k_gemm_wmma<64><<<M/64, 128, 8192, stream>>>(T16, W2_16, T1, 4, 64, 64, 64, 0);
  k_gather<<<CDIV((int)E,256),256,0,stream>>>(T1, VAL, IDX, gB2, T2);
  k_groupnorm_silu<<<Bc*4,256,0,stream>>>(T2, gn2G, gn2B, GCN, 0);

  // ---- row mamba (seqs = B*H = 1024, L = 32, di = 128) ----
  k_bcn2bnc_bf16<<<CDIV((int)E,256),256,0,stream>>>(x, MX16);
  k_gemm_wmma<64><<<M/64,128,32768,stream>>>(MX16, rWin16, XZ, 16, 256, 64, 256, 1);
  k_conv_silu<<<CDIV(M*128,256),256,0,stream>>>(XZ, rCw, rCb, XC, XC16, 1024, 32, 128, 256);
  k_gemm_wmma<128><<<M/64,128,12288,stream>>>(XC16, rWx16, DBC, 3, 36, 128, 48, 1);
  k_dt<<<CDIV(M*128,256),256,0,stream>>>(DBC, rWdt, rBdt, DT, M, 128, 4, 48);
  k_scan<<<CDIV(1024*128,256),256,0,stream>>>(DT, XC, DBC, XZ, rAl, rD, Yb, Y16,
                                              1024, 32, 128, 48, 4, 256);
  k_gemm_wmma<128><<<M/64,128,16384,stream>>>(Y16, rWo16, MOUT, 4, 64, 128, 64, 1);

  // ---- col mamba (transpose H<->W, seqs = B*W = 1024, L = 32) ----
  k_row2col_bf16<<<CDIV((int)E,256),256,0,stream>>>(MOUT, MX16);
  k_gemm_wmma<64><<<M/64,128,32768,stream>>>(MX16, cWin16, XZ, 16, 256, 64, 256, 1);
  k_conv_silu<<<CDIV(M*128,256),256,0,stream>>>(XZ, cCw, cCb, XC, XC16, 1024, 32, 128, 256);
  k_gemm_wmma<128><<<M/64,128,12288,stream>>>(XC16, cWx16, DBC, 3, 36, 128, 48, 1);
  k_dt<<<CDIV(M*128,256),256,0,stream>>>(DBC, cWdt, cBdt, DT, M, 128, 4, 48);
  k_scan<<<CDIV(1024*128,256),256,0,stream>>>(DT, XC, DBC, XZ, cAl, cD, Yb, Y16,
                                              1024, 32, 128, 48, 4, 256);
  k_gemm_wmma<128><<<M/64,128,16384,stream>>>(Y16, cWo16, MOUT, 4, 64, 128, 64, 1);
  k_groupnorm_silu<<<Bc*4,256,0,stream>>>(MOUT, spaG, spaB, T1, 2);   // spa

  // ---- spectral mamba (seqs = B*H*W = 32768, L = 8, di = 16, r = 1) ----
  const long M2 = (long)Bc*Nn*8;            // 262144 rows
  k_sp_xz<<<CDIV((int)(M2*32),256),256,0,stream>>>(x, sPe, sWin, XZ);
  k_conv_silu<<<CDIV((int)(M2*16),256),256,0,stream>>>(XZ, sCw, sCb, XC, XC16,
                                                       32768, 8, 16, 32);
  k_sgemm_xwT<<<CDIV((int)(M2*33),256),256,0,stream>>>(XC, sWx, DBC, M2, 33, 16, 33);
  k_dt<<<CDIV((int)(M2*16),256),256,0,stream>>>(DBC, sWdt, sBdt, DT, M2, 16, 1, 33);
  k_scan<<<CDIV(32768*16,256),256,0,stream>>>(DT, XC, DBC, XZ, sAl, sD, Yb, Y16,
                                              32768, 8, 16, 33, 1, 32);
  k_sgemm_xwT<<<CDIV((int)(M2*8),256),256,0,stream>>>(Yb, sWo, MOUT, M2, 8, 16, 8);
  k_groupnorm_silu<<<Bc*4,256,0,stream>>>(MOUT, speG, speB, T2, 0);   // spe

  // ---- fusion ----
  k_fusion<<<CDIV((int)E,256),256,0,stream>>>(GCN, T1, T2, x, bothW, fusW, out);
}
